// Mean_Average_Precision_19980187860999
// MI455X (gfx1250) — matrix operand order repaired and checked by reading, not compiled, not run on hardware
//
#include <hip/hip_runtime.h>
#include <hip/hip_bf16.h>
#include <stdint.h>

#define NCLS 80
#define NPRED 30000
#define NGT 800
#define NSORT 32768

typedef _Float16 v16h __attribute__((ext_vector_type(16)));
typedef float    v8f  __attribute__((ext_vector_type(8)));

// ---------------------------------------------------------------- init first[]
__global__ void k_init_first(int* first) {
  int i = blockIdx.x * 256 + threadIdx.x;
  if (i < NCLS * NGT) first[i] = NPRED;
}

// ---------------------------------------------------------------- per-class sort
// One workgroup per class. Bitonic sort of 32768 64-bit keys in 256KB dynamic LDS
// (CDNA5: up to 320KB LDS per workgroup). Key = score_bits<<32 | ~idx, sorted
// descending => stable descending score order, exactly like argsort(-score).
// Pair-index enumeration: every lane does a compare-exchange every iteration
// (no wasted ixj<i half), 16 fully-active iterations per pass.
__global__ __launch_bounds__(1024) void k_sort(const float* __restrict__ pred,
                                               int* __restrict__ sorted_idx) {
  extern __shared__ unsigned long long key[];
  const int c   = blockIdx.x;
  const int tid = threadIdx.x;
  const float* pb = pred + (size_t)c * NPRED * 7;
  for (int i = tid; i < NSORT; i += 1024) {
    unsigned long long k = 0ull;
    if (i < NPRED) {
      unsigned sb = __float_as_uint(pb[(size_t)i * 7 + 2]);   // score in [0,1): bits monotonic
      k = ((unsigned long long)sb << 32) | (unsigned)(~(unsigned)i);
    }
    key[i] = k;
  }
  __syncthreads();
  const int NPAIR = NSORT / 2;  // 16384 compare-exchanges per pass
  for (int kk = 2; kk <= NSORT; kk <<= 1) {
    for (int j = kk >> 1; j > 0; j >>= 1) {
      for (int p = tid; p < NPAIR; p += 1024) {
        int i  = ((p & ~(j - 1)) << 1) | (p & (j - 1));  // index with bit j clear
        int ip = i | j;                                   // partner
        unsigned long long a = key[i];
        unsigned long long b = key[ip];
        bool desc = ((i & kk) == 0);
        if (desc ? (a < b) : (a > b)) { key[i] = b; key[ip] = a; }
      }
      __syncthreads();
    }
  }
  for (int r = tid; r < NPRED; r += 1024)
    sorted_idx[(size_t)c * NPRED + r] = (int)(~(unsigned)key[r]);
}

// ---------------------------------------------------------------- match kernel
// Stage 800 GT boxes (cols 3..6) into LDS with CDNA5 async-to-LDS loads
// (ASYNCcnt path), then each thread computes 800 IoUs for one sorted prediction.
__global__ __launch_bounds__(256) void k_match(const float* __restrict__ pred,
                                               const float* __restrict__ gt,
                                               const int* __restrict__ sorted_idx,
                                               int* __restrict__ match,
                                               int* __restrict__ first) {
  __shared__ __align__(16) float g4[NGT * 4];
  __shared__ float garea[NGT];
  const int c   = blockIdx.y;
  const int tid = threadIdx.x;
  const float* gbase = gt + (size_t)c * NGT * 7;

  for (int e = tid; e < NGT * 4; e += 256) {
    int gi = e >> 2, cc = e & 3;
    const float* gp = gbase + (size_t)gi * 7 + 3 + cc;
    unsigned loff = (unsigned)(uintptr_t)(&g4[e]);     // low 32 bits = LDS byte offset
    asm volatile("global_load_async_to_lds_b32 %0, %1, off"
                 :: "v"(loff), "v"(gp) : "memory");
  }
  asm volatile("s_wait_asynccnt 0x0" ::: "memory");
  __syncthreads();

  for (int gi = tid; gi < NGT; gi += 256) {
    float x1 = g4[gi*4+0], y1 = g4[gi*4+1], x2 = g4[gi*4+2], y2 = g4[gi*4+3];
    garea[gi] = (x2 - x1) * (y2 - y1);
  }
  __syncthreads();

  const int r = blockIdx.x * 256 + tid;
  if (r >= NPRED) return;
  const int idx = sorted_idx[(size_t)c * NPRED + r];
  const float* p = pred + ((size_t)c * NPRED + (size_t)idx) * 7;
  float ax1 = p[3], ay1 = p[4], ax2 = p[5], ay2 = p[6];
  float aarea = (ax2 - ax1) * (ay2 - ay1);

  float best  = -1.0f;   // strict '>' keeps FIRST max (jnp.argmax semantics)
  int   bestg = 0;
  #pragma unroll 4
  for (int g = 0; g < NGT; ++g) {
    float4 b = *(const float4*)&g4[g * 4];
    float lx = fmaxf(ax1, b.x), ly = fmaxf(ay1, b.y);
    float rx = fminf(ax2, b.z), ry = fminf(ay2, b.w);
    float w = fmaxf(rx - lx, 0.0f), h = fmaxf(ry - ly, 0.0f);
    float inter = w * h;
    float iou = inter / (aarea + garea[g] - inter + 1e-9f);
    if (iou > best) { best = iou; bestg = g; }
  }
  bool valid = best > 0.5f;
  match[(size_t)c * NPRED + r] = valid ? bestg : -1;
  if (valid) atomicMin(&first[c * NGT + bestg], r);     // order-independent => deterministic
}

// ---------------------------------------------------------------- AP kernel
// One wave32 per class. TP prefix-sums via v_wmma_f32_16x16x32_f16:
// X is a 16x32 f16 tile of TP flags (linear i = M*32 + k), B[k][n] = (k <= 2n+1),
// so D[M][n] = inclusive prefix of row M through element 2n+1 (f16 exact for 0/1,
// f32 accumulate => exact integer cumsums, identical to the reference).
__global__ __launch_bounds__(32) void k_ap(const int* __restrict__ match,
                                           const int* __restrict__ first,
                                           float* __restrict__ ap) {
  __shared__ float xt[512];
  __shared__ float rs[16];
  __shared__ float red[32];
  const int c    = blockIdx.x;
  const int lane = threadIdx.x;
  const int half = lane >> 4;
  const int lm   = lane & 15;
  const int* mc = match + (size_t)c * NPRED;
  const int* fc = first + c * NGT;

  // B (32x16, f16): element h<8: K=h+8*half, h>=8: K=16+(h-8)+8*half.
  v16h B;
  #pragma unroll
  for (int h = 0; h < 16; ++h) {
    int K = (h < 8) ? (h + 8 * half) : (16 + (h - 8) + 8 * half);
    B[h] = (_Float16)((K <= 2 * lm + 1) ? 1.0f : 0.0f);
  }

  float apacc = 0.0f, running = 0.0f;
  const int T = (NPRED + 511) / 512;
  for (int t = 0; t < T; ++t) {
    const int tb = t * 512;
    for (int e = lane; e < 512; e += 32) {
      int i = tb + e;
      float v = 0.0f;
      if (i < NPRED) {
        int mg = mc[i];
        if (mg >= 0 && fc[mg] == i) v = 1.0f;     // tp flag
      }
      xt[e] = v;
    }
    __syncthreads();

    // A (16x32 f16): lane M=lm; VGPR v<4: K=2v+e+8*half, v>=4: K=16+2(v-4)+e+8*half.
    v16h A;
    #pragma unroll
    for (int h = 0; h < 16; ++h) {
      int v = h >> 1, e = h & 1;
      int K = (v < 4) ? (2 * v + e + 8 * half) : (16 + 2 * (v - 4) + e + 8 * half);
      A[h] = (_Float16)xt[lm * 32 + K];
    }
    v8f Cz = {0.f, 0.f, 0.f, 0.f, 0.f, 0.f, 0.f, 0.f};
    v8f D = __builtin_amdgcn_wmma_f32_16x16x32_f16(false, A, false, B,
                                                   (short)0, Cz, false, false);

    // row sums (N==15 => prefix through k=31) -> LDS for cross-row offsets
    if (lm == 15) {
      #pragma unroll
      for (int rr = 0; rr < 8; ++rr) rs[rr + 8 * half] = D[rr];
    }
    __syncthreads();
    float pre[17];
    pre[0] = 0.0f;
    #pragma unroll
    for (int m = 0; m < 16; ++m) pre[m + 1] = pre[m] + rs[m];

    // Each lane owns elements (M = rr+8*half, k = 2*lm, 2*lm+1).
    // prec[i] = ctp[i]/((i+1)+eps) since ctp+cfp == i+1 exactly.
    #pragma unroll
    for (int rr = 0; rr < 8; ++rr) {
      int M  = rr + 8 * half;
      int k1 = 2 * lm + 1;
      float tp1 = xt[M * 32 + k1];
      float tp0 = xt[M * 32 + k1 - 1];
      float P1 = running + pre[M] + D[rr];   // inclusive ctp at element k1
      float P0 = P1 - tp1;                   // inclusive ctp at element k1-1
      int i1 = tb + M * 32 + k1;
      int i0 = i1 - 1;
      if (i0 >= 1 && i0 < NPRED)
        apacc += tp0 * (P0 / ((float)(i0 + 1) + 1e-9f) + (P0 - tp0) / ((float)i0 + 1e-9f));
      if (i1 >= 1 && i1 < NPRED)
        apacc += tp1 * (P1 / ((float)(i1 + 1) + 1e-9f) + (P1 - tp1) / ((float)i1 + 1e-9f));
    }
    running += pre[16];
    __syncthreads();
  }

  red[lane] = apacc;
  __syncthreads();
  if (lane == 0) {
    float s = 0.0f;
    for (int i = 0; i < 32; ++i) s += red[i];
    ap[c] = s * 0.5f / (800.0f + 1e-9f);
  }
}

// ---------------------------------------------------------------- mean over classes
__global__ __launch_bounds__(128) void k_mean(const float* __restrict__ ap,
                                              float* __restrict__ out) {
  __shared__ float sh[128];
  int t = threadIdx.x;
  sh[t] = (t < NCLS) ? ap[t] : 0.0f;
  __syncthreads();
  for (int s = 64; s > 0; s >>= 1) {
    if (t < s) sh[t] += sh[t + s];
    __syncthreads();
  }
  if (t == 0) out[0] = sh[0] / (float)NCLS;
}

// ---------------------------------------------------------------- launch
extern "C" void kernel_launch(void* const* d_in, const int* in_sizes, int n_in,
                              void* d_out, int out_size, void* d_ws, size_t ws_size,
                              hipStream_t stream) {
  const float* pred = (const float*)d_in[0];   // (80,30000,7) f32
  const float* gt   = (const float*)d_in[1];   // (80,800,7)   f32
  char* ws = (char*)d_ws;
  int*   sorted_idx = (int*)ws;                                            // 9.6 MB
  int*   match      = (int*)(ws + (size_t)NCLS * NPRED * sizeof(int));     // 9.6 MB
  int*   first      = (int*)(ws + 2ull * NCLS * NPRED * sizeof(int));      // 256 KB
  float* ap         = (float*)(ws + 2ull * NCLS * NPRED * sizeof(int)
                                  + (size_t)NCLS * NGT * sizeof(int));     // 320 B

  (void)hipFuncSetAttribute((const void*)k_sort,
                            hipFuncAttributeMaxDynamicSharedMemorySize,
                            NSORT * (int)sizeof(unsigned long long));

  k_init_first<<<(NCLS * NGT + 255) / 256, 256, 0, stream>>>(first);
  k_sort<<<NCLS, 1024, NSORT * sizeof(unsigned long long), stream>>>(pred, sorted_idx);
  dim3 mgrid((NPRED + 255) / 256, NCLS);
  k_match<<<mgrid, 256, 0, stream>>>(pred, gt, sorted_idx, match, first);
  k_ap<<<NCLS, 32, 0, stream>>>(match, first, ap);
  k_mean<<<1, 128, 0, stream>>>(ap, (float*)d_out);
}
